// Generator_75041668595885
// MI455X (gfx1250) — matrix operand order repaired and checked by reading, not compiled, not run on hardware
//
#include <hip/hip_runtime.h>
#include <cstdint>

// Shapes from the reference
#define B_     64
#define S_     256
#define VOCAB_ 2048
#define EMB_   256
#define HID_   1024
#define GATES_ 4096   // 4*HID

typedef __attribute__((ext_vector_type(16))) __bf16 v16bf;
typedef __attribute__((ext_vector_type(8)))  float  v8f;
typedef unsigned int u32x4 __attribute__((ext_vector_type(4)));
typedef int          i32x4 __attribute__((ext_vector_type(4)));
typedef int          i32x8 __attribute__((ext_vector_type(8)));

union ABv { unsigned int u[8]; v16bf v; };

// LDS B panel: 64 rows x 32 halfs, padded to 40 halfs/row (80B: 16B-aligned,
// 16 distinct banks across the 16 lanes of a B-tile read).
#define PANEL_ROW_HALFS 40
#define PANEL_BYTES     (64 * PANEL_ROW_HALFS * 2)   // 5120

#if defined(__AMDGCN__) && __has_builtin(__builtin_amdgcn_tensor_load_to_lds)
#define HAVE_TDM 1
#else
#define HAVE_TDM 0
#endif

__device__ __forceinline__ unsigned short f2bf(float f) {
    unsigned int u = __float_as_uint(f);
    unsigned int r = u + 0x7FFFu + ((u >> 16) & 1u);   // round-to-nearest-even
    return (unsigned short)(r >> 16);
}

// Issue one TDM DMA: 2D tile 32(K) x 64(N) bf16 from WT[n0.., k0..] into LDS
// at byte offset ldsOff, with LDS padding 64B-row -> +16B (stride 80B).
__device__ __forceinline__ void tdm_load_panel(const unsigned short* g,
                                               unsigned ldsOff, int ldwElems) {
#if HAVE_TDM
    unsigned long long ga = (unsigned long long)(size_t)g;
    u32x4 g0 = {};
    g0[0] = 1u;                                   // count=1 (valid user D#)
    g0[1] = ldsOff;                               // lds_addr (bytes)
    g0[2] = (unsigned)(ga & 0xFFFFFFFFull);       // global_addr[31:0]
    g0[3] = (unsigned)((ga >> 32) & 0x01FFFFFFull) | (2u << 30);  // addr[56:32], type=2
    i32x8 g1 = {};
    // data_size=2B (code1), pad_enable, pad_interval=16DW (code3), pad_amount=4DW (code3)
    g1[0] = (1 << 16) | (1 << 20) | (3 << 22) | (3 << 25);
    g1[1] = (int)(32u << 16);                     // tensor_dim0 = 32  (bits 79:48)
    g1[2] = (int)(64u << 16);                     // tensor_dim1 = 64  (bits 111:80)
    g1[3] = (int)(32u << 16);                     // tile_dim0   = 32  (bits 127:112)
    g1[4] = 64;                                   // tile_dim1   = 64
    g1[5] = ldwElems;                             // tensor_dim0_stride[31:0]
    g1[6] = 0; g1[7] = 0;
    i32x4 g2 = {}, g3 = {};
#if __clang_major__ >= 23
    i32x8 g4 = {};
    __builtin_amdgcn_tensor_load_to_lds(g0, g1, g2, g3, g4, 0);
#else
    __builtin_amdgcn_tensor_load_to_lds(g0, g1, g2, g3, 0);
#endif
#endif
}

__device__ __forceinline__ void tdm_wait0() {
#if defined(__AMDGCN__) && __has_builtin(__builtin_amdgcn_s_wait_tensorcnt)
    __builtin_amdgcn_s_wait_tensorcnt(0);
#else
    asm volatile("s_wait_tensorcnt 0x0" ::: "memory");
#endif
}

// The TDM engine writes LDS through an SGPR descriptor, invisibly to the
// optimizer. Escape the panel pointer into an asm with a memory clobber so
// the subsequent ds_load_b128 reads cannot be folded to undef / hoisted.
__device__ __forceinline__ void lds_publish(const void* p) {
    asm volatile("" : "+v"(p) : : "memory");
}

// ---------------------------------------------------------------- utility
__global__ __launch_bounds__(256) void zero_kernel(unsigned int* p, int n) {
    int i = blockIdx.x * blockDim.x + threadIdx.x;
    if (i < n) p[i] = 0u;
}

// dst[n*ldd + koff + k] = bf16(src[k*N + n])   (store W^T in bf16)
__global__ __launch_bounds__(256) void transpose_bf16_kernel(
    const float* __restrict__ src, unsigned short* __restrict__ dst,
    int K, int N, int ldd, int koff)
{
    int i = blockIdx.x * blockDim.x + threadIdx.x;
    if (i >= K * N) return;
    int k = i / N, n = i % N;
    dst[(size_t)n * ldd + koff + k] = f2bf(src[(size_t)k * N + n]);
}

// Xe[((t*B)+b)*EMB + e] = bf16(emb[x[b,t]*EMB + e])  (time-major embedded input)
__global__ __launch_bounds__(256) void embed_kernel(
    const int* __restrict__ x, const float* __restrict__ emb,
    unsigned short* __restrict__ Xe)
{
    int i  = blockIdx.x * blockDim.x + threadIdx.x;   // < S*B*EMB
    int e  = i & (EMB_ - 1);
    int bt = i >> 8;            // EMB = 256
    int b  = bt & (B_ - 1);
    int t  = bt >> 6;           // B = 64
    int tok = x[b * S_ + t];
    Xe[i] = f2bf(emb[(size_t)tok * EMB_ + e]);
}

// ---------------------------------------------------------------- WMMA GEMM
// C[M,N] = concatK(A0[M,K0], A1[M,Ktot-K0]) x W^T  (+bias, optional sigmoid)
// 4 waves/WG: waves share one 64-column B panel (TDM double-buffered in LDS),
// each wave owns one 16-row M tile; 4 WMMA accumulators per wave.
// A loads are software-pipelined one K-step ahead; all 4 B tiles are loaded
// into registers before the WMMA chain so ds latency overlaps the matrix pipe.
__global__ __launch_bounds__(128) void wmma_gemm_kernel(
    const unsigned short* __restrict__ A0, int lda0, int K0,
    const unsigned short* __restrict__ A1, int lda1,
    const unsigned short* __restrict__ WT, int ldw, int Ktot,
    int numM, int N,
    const float* __restrict__ bias,
    float* __restrict__ outF, unsigned short* __restrict__ outB,
    int ldo, int do_sigmoid)
{
    __shared__ unsigned short Bpanel[2][64 * PANEL_ROW_HALFS];

    const int lane = threadIdx.x & 31;
    const int wid  = threadIdx.x >> 5;
    const int w    = (blockIdx.x << 2) + wid;
    const int mt   = w % numM;
    const int nq   = w / numM;          // identical for all 4 waves of a block
    const int m0   = mt << 4;
    const int n0   = nq << 6;
    if (n0 >= N) return;                // uniform per block (grids divide exactly)

    v8f acc[4] = {};

    // A layout (16-bit A 16x32): lane<16 -> K 0..7 & 16..23 ; lane>=16 -> +8
    const int arow = m0 + (lane & 15);
    const int kbA  = (lane & 16) ? 8 : 0;
    // B layout (16-bit B 32x16): lane = N row of W^T; lane<16 -> K 0..7 & 16..23 ; lane>=16 -> +8
    const int nlow = lane & 15;
    const int kbB  = (lane & 16) ? 8 : 0;

    const unsigned short* a0base = A0 + (size_t)arow * lda0 + kbA;
    const unsigned short* a1base = A1 + (size_t)arow * lda1 + kbA;

#if HAVE_TDM
    if (wid == 0)
        tdm_load_panel(WT + (size_t)n0 * ldw, 0u, ldw);
#endif

    // ---- prologue: A tile for k0 = 0
    uint4 alo, ahi;
    {
        const unsigned short* ap = (0 < K0) ? a0base : (a1base - K0);
        alo = *(const uint4*)(ap);
        ahi = *(const uint4*)(ap + 16);
    }

    for (int k0 = 0; k0 < Ktot; k0 += 32) {
        const int buf = (k0 >> 5) & 1;

#if HAVE_TDM
        if (wid == 0) tdm_wait0();      // this buffer's DMA complete
        __syncthreads();                // publish panel; prior reads of other buf done
        lds_publish(&Bpanel[0][0]);     // TDM wrote LDS: keep ds loads alive
        if (wid == 0 && (k0 + 32) < Ktot)
            tdm_load_panel(WT + (size_t)n0 * ldw + (k0 + 32),
                           (unsigned)((buf ^ 1) * PANEL_BYTES), ldw);
#else
        __syncthreads();                // prior reads of this buffer done
        {
            int r = threadIdx.x >> 1;
            int c = (threadIdx.x & 1) << 4;
            const unsigned short* src = WT + (size_t)(n0 + r) * ldw + k0 + c;
            unsigned short* dst = &Bpanel[buf][r * PANEL_ROW_HALFS + c];
            *(uint4*)dst       = *(const uint4*)src;
            *((uint4*)dst + 1) = *((const uint4*)src + 1);
        }
        __syncthreads();
#endif

        // ---- prefetch A tile for next K-step (overlaps with WMMA below)
        uint4 nlo = {}, nhi = {};
        const int kn = k0 + 32;
        if (kn < Ktot) {
            const unsigned short* ap =
                (kn < K0) ? (a0base + kn) : (a1base + (kn - K0));
            nlo = *(const uint4*)(ap);
            nhi = *(const uint4*)(ap + 16);
        }

        ABv a;
        a.u[0] = alo.x; a.u[1] = alo.y; a.u[2] = alo.z; a.u[3] = alo.w;
        a.u[4] = ahi.x; a.u[5] = ahi.y; a.u[6] = ahi.z; a.u[7] = ahi.w;

        // ---- load all 4 B tiles from LDS first (independent registers),
        // then run the WMMA chain: ds latency overlaps matrix pipe.
        ABv b[4];
#pragma unroll
        for (int q = 0; q < 4; ++q) {
            const unsigned short* bp =
                &Bpanel[buf][(unsigned)((q << 4) + nlow) * PANEL_ROW_HALFS + kbB];
            uint4 blo = *(const uint4*)bp;
            uint4 bhi = *(const uint4*)(bp + 16);
            b[q].u[0] = blo.x; b[q].u[1] = blo.y; b[q].u[2] = blo.z; b[q].u[3] = blo.w;
            b[q].u[4] = bhi.x; b[q].u[5] = bhi.y; b[q].u[6] = bhi.z; b[q].u[7] = bhi.w;
        }
#pragma unroll
        for (int q = 0; q < 4; ++q) {
            acc[q] = __builtin_amdgcn_wmma_f32_16x16x32_bf16(
                false, a.v, false, b[q].v, (short)0, acc[q], false, false);
        }

        alo = nlo; ahi = nhi;           // rotate pipelined A registers
    }

    // C/D layout: VGPR r, lanes 0-15 -> M=r, lanes 16-31 -> M=r+8; N = lane&15
    const int rbase = (lane & 16) ? 8 : 0;
#pragma unroll
    for (int q = 0; q < 4; ++q) {
        const int col = n0 + (q << 4) + nlow;
        const float bv = bias ? bias[col] : 0.0f;
#pragma unroll
        for (int r = 0; r < 8; ++r) {
            const int row = m0 + rbase + r;
            float v = acc[q][r] + bv;
            if (do_sigmoid) v = 1.0f / (1.0f + __expf(-v));
            if (outF) outF[(size_t)row * ldo + col] = v;
            else      outB[(size_t)row * ldo + col] = f2bf(v);
        }
    }
}

// ---------------------------------------------------------------- LSTM gates
__global__ __launch_bounds__(256) void lstm_gates_kernel(
    const float* __restrict__ Z, const float* __restrict__ bias,
    float* __restrict__ Hf, float* __restrict__ Cf,
    unsigned short* __restrict__ Hb,
    const int* __restrict__ x, int t, int isL2,
    float* __restrict__ OutF, unsigned short* __restrict__ seqb)
{
    int idx = blockIdx.x * blockDim.x + threadIdx.x;  // < B*HID
    int b = idx >> 10;
    int j = idx & (HID_ - 1);
    const float* zr = Z + (size_t)b * GATES_;

    float zi = zr[j]          + bias[j];
    float zf = zr[HID_   + j] + bias[HID_   + j];
    float zg = zr[2*HID_ + j] + bias[2*HID_ + j];
    float zo = zr[3*HID_ + j] + bias[3*HID_ + j];

    float ig = 1.0f / (1.0f + __expf(-zi));
    float fg = 1.0f / (1.0f + __expf(-zf));
    float og = 1.0f / (1.0f + __expf(-zo));
    float g  = tanhf(zg);

    float cold = Cf[idx], hold = Hf[idx];
    float cn = fg * cold + ig * g;
    float hn = og * tanhf(cn);

    bool m = (x[b * S_ + t] != 0);
    float cw = m ? cn : cold;
    float hw = m ? hn : hold;

    Cf[idx] = cw;
    Hf[idx] = hw;
    Hb[idx] = f2bf(hw);

    if (isL2) {
        float po  = OutF[idx];
        float out = m ? hw : po;           // where(m, h2_new, prev_out)
        OutF[idx] = out;
        seqb[(size_t)(b * S_ + t) * HID_ + j] = f2bf(out);
    }
}

// ---------------------------------------------------------------- launcher
extern "C" void kernel_launch(void* const* d_in, const int* in_sizes, int n_in,
                              void* d_out, int out_size, void* d_ws, size_t ws_size,
                              hipStream_t stream)
{
    const int*   x   = (const int*)  d_in[0];
    const float* emb = (const float*)d_in[1];
    const float* W1  = (const float*)d_in[2];
    const float* U1  = (const float*)d_in[3];
    const float* b1  = (const float*)d_in[4];
    const float* W2  = (const float*)d_in[5];
    const float* U2  = (const float*)d_in[6];
    const float* b2  = (const float*)d_in[7];
    const float* Wd1 = (const float*)d_in[8];
    const float* bd1 = (const float*)d_in[9];
    const float* Wd2 = (const float*)d_in[10];
    const float* bd2 = (const float*)d_in[11];
    float* out = (float*)d_out;

    // ---- workspace layout (256B aligned slabs)
    uint8_t* ws = (uint8_t*)d_ws;
    size_t off = 0;
    auto alloc = [&](size_t bytes) -> void* {
        void* p = ws + off;
        off += (bytes + 255) & ~(size_t)255;
        return p;
    };
    unsigned short* WT1  = (unsigned short*)alloc((size_t)GATES_ * 1280 * 2); // (4096 x 1280)
    unsigned short* WT2  = (unsigned short*)alloc((size_t)GATES_ * 2048 * 2); // (4096 x 2048)
    unsigned short* WdT1 = (unsigned short*)alloc((size_t)HID_   * 1024 * 2); // (1024 x 1024)
    unsigned short* WdT2 = (unsigned short*)alloc((size_t)VOCAB_ * 1024 * 2); // (2048 x 1024)
    unsigned short* Xe   = (unsigned short*)alloc((size_t)S_ * B_ * EMB_ * 2);
    float* Z1   = (float*)alloc((size_t)B_ * GATES_ * 4);
    float* Z2   = (float*)alloc((size_t)B_ * GATES_ * 4);
    // contiguous state block (zeroed together)
    float* H1f  = (float*)alloc((size_t)B_ * HID_ * 4);
    float* C1   = (float*)alloc((size_t)B_ * HID_ * 4);
    float* H2f  = (float*)alloc((size_t)B_ * HID_ * 4);
    float* C2   = (float*)alloc((size_t)B_ * HID_ * 4);
    float* OutF = (float*)alloc((size_t)B_ * HID_ * 4);
    unsigned short* H1b = (unsigned short*)alloc((size_t)B_ * HID_ * 2);
    unsigned short* H2b = (unsigned short*)alloc((size_t)B_ * HID_ * 2);
    unsigned short* seqb = (unsigned short*)alloc((size_t)B_ * S_ * HID_ * 2);
    unsigned short* hb   = (unsigned short*)alloc((size_t)B_ * S_ * HID_ * 2);

    // ---- weight conversion (fp32 -> bf16, transposed)
    transpose_bf16_kernel<<<(EMB_ * GATES_) / 256, 256, 0, stream>>>(W1,  WT1,  EMB_, GATES_, 1280, 0);
    transpose_bf16_kernel<<<(HID_ * GATES_) / 256, 256, 0, stream>>>(U1,  WT1,  HID_, GATES_, 1280, 256);
    transpose_bf16_kernel<<<(HID_ * GATES_) / 256, 256, 0, stream>>>(W2,  WT2,  HID_, GATES_, 2048, 0);
    transpose_bf16_kernel<<<(HID_ * GATES_) / 256, 256, 0, stream>>>(U2,  WT2,  HID_, GATES_, 2048, 1024);
    transpose_bf16_kernel<<<(HID_ * HID_)   / 256, 256, 0, stream>>>(Wd1, WdT1, HID_, HID_,   1024, 0);
    transpose_bf16_kernel<<<(HID_ * VOCAB_) / 256, 256, 0, stream>>>(Wd2, WdT2, HID_, VOCAB_, 1024, 0);

    // ---- embedding (time-major bf16)
    embed_kernel<<<(S_ * B_ * EMB_) / 256, 256, 0, stream>>>(x, emb, Xe);

    // ---- zero recurrent state (H1f..H2b contiguous)
    {
        int ndw = (5 * B_ * HID_ * 4 + 2 * B_ * HID_ * 2) / 4;
        zero_kernel<<<(ndw + 255) / 256, 256, 0, stream>>>((unsigned int*)H1f, ndw);
    }

    // ---- sequential LSTM: 2 GEMM + 2 gate kernels per step
    const int recBlocks = (4 /*numM*/ * (GATES_ / 64)) / 4;   // 64 WGs of 128 thr
    for (int t = 0; t < S_; ++t) {
        // layer 1: z1 = [x_t ; h1] @ [W1;U1]
        wmma_gemm_kernel<<<recBlocks, 128, 0, stream>>>(
            Xe + (size_t)t * B_ * EMB_, EMB_, EMB_,
            H1b, HID_, WT1, 1280, 1280,
            B_ / 16, GATES_, nullptr, Z1, nullptr, GATES_, 0);
        lstm_gates_kernel<<<(B_ * HID_) / 256, 256, 0, stream>>>(
            Z1, b1, H1f, C1, H1b, x, t, 0, nullptr, nullptr);
        // layer 2: z2 = [h1 ; h2] @ [W2;U2]
        wmma_gemm_kernel<<<recBlocks, 128, 0, stream>>>(
            H1b, HID_, HID_,
            H2b, HID_, WT2, 2048, 2048,
            B_ / 16, GATES_, nullptr, Z2, nullptr, GATES_, 0);
        lstm_gates_kernel<<<(B_ * HID_) / 256, 256, 0, stream>>>(
            Z2, b2, H2f, C2, H2b, x, t, 1, OutF, seqb);
    }

    // ---- decode layer 1: h = sigmoid(seq @ Wd1 + bd1) -> bf16
    {
        int numM = (B_ * S_) / 16;                 // 1024
        int blocks = (numM * (HID_ / 64)) / 4;     // 4096
        wmma_gemm_kernel<<<blocks, 128, 0, stream>>>(
            seqb, HID_, 1 << 30, seqb, HID_,
            WdT1, HID_, HID_,
            numM, HID_, bd1, nullptr, hb, HID_, 1);
    }
    // ---- decode layer 2: out = sigmoid(h @ Wd2 + bd2) -> fp32 d_out
    {
        int numM = (B_ * S_) / 16;                 // 1024
        int blocks = (numM * (VOCAB_ / 64)) / 4;   // 8192
        wmma_gemm_kernel<<<blocks, 128, 0, stream>>>(
            hb, HID_, 1 << 30, hb, HID_,
            WdT2, HID_, HID_,
            numM, VOCAB_, bd2, out, nullptr, VOCAB_, 1);
    }
}
